// HypAttention_39496519254613
// MI455X (gfx1250) — matrix operand order repaired and checked by reading, not compiled, not run on hardware
//
#include <hip/hip_runtime.h>
#include <hip/hip_bf16.h>

// ---------------------------------------------------------------------------
// HypAttention for MI455X (gfx1250): bf16 WMMA for all GEMM-shaped stages,
// f32 everywhere the fp() feature map / normalizations need precision.
// 32-row M-blocking: each B fragment fetched from L2 feeds two WMMAs.
// ---------------------------------------------------------------------------

#define NROWS   100000
#define IN_C    256
#define OUT_C   64
#define NH      8
#define QKV_COLS 1536           // q(512) | k(512) | v(512)
#define CT_QKV  96              // 1536 / 16 column tiles
#define KK_QKV  8               // 256 / 32 K slices
#define ROWT    32              // rows per workgroup (2 x 16 sub-tiles)

typedef __attribute__((ext_vector_type(16))) __bf16 v16bf;
typedef __attribute__((ext_vector_type(8)))  float  v8f;

struct __align__(16) B128 { unsigned int u[4]; };
union ABu { v16bf v; B128 b[2]; };

__device__ __forceinline__ unsigned short f2bf(float f) {
    union { float f; unsigned int u; } v; v.f = f;
    unsigned int u = v.u;
    unsigned int r = (u + 0x7FFFu + ((u >> 16) & 1u)) >> 16;   // RNE
    return (unsigned short)r;
}
__device__ __forceinline__ float bf2f(unsigned short h) {
    union { unsigned int u; float f; } v; v.u = ((unsigned int)h) << 16;
    return v.f;
}

// ---------------------------------------------------------------------------
// Kernel 1: pack Wq/Wk/Wv -> fused bf16 B-operand layout, biases, Wm.
// B layout for V_WMMA_F32_16X16X32_BF16 (32x16, K x N), contiguous k order used here (A matches):
//   lanes 0-15: n=lane, elements 0..15 = K 0..15 ; lanes 16-31: K 16..31.
// Packed so each lane's 16 bf16 are contiguous (32B): [ct][kk][lane][16]
// ---------------------------------------------------------------------------
__global__ void prep_pack(const float* __restrict__ Wq, const float* __restrict__ bq,
                          const float* __restrict__ Wk, const float* __restrict__ bk,
                          const float* __restrict__ Wv, const float* __restrict__ bv,
                          const float* __restrict__ Wm,
                          unsigned short* __restrict__ Bqkv,
                          float* __restrict__ biasQKV,
                          unsigned short* __restrict__ Bwm)
{
    int i = blockIdx.x * 256 + threadIdx.x;
    const int NQKV = CT_QKV * KK_QKV * 32 * 16;   // 393216
    if (i < NQKV) {
        int j  = i & 15, l = (i >> 4) & 31, kk = (i >> 9) & 7, ct = i >> 12;
        int n  = l & 15;
        int kb = (l < 16) ? 0 : 16;
        int K  = kk * 32 + kb + j;
        int col = ct * 16 + n;
        float w;
        if (col < 512)       { int h = col >> 6, o = col & 63;
                               w = Wq[((size_t)h * IN_C + K) * OUT_C + o]; }
        else if (col < 1024) { int c = col - 512; int h = c >> 6, o = c & 63;
                               w = Wk[((size_t)h * IN_C + K) * OUT_C + o]; }
        else                 { int c = col - 1024; int h = c >> 6, o = c & 63;
                               w = Wv[((size_t)h * IN_C + K) * OUT_C + o]; }
        Bqkv[i] = f2bf(w);
        return;
    }
    i -= NQKV;
    if (i < QKV_COLS) {
        biasQKV[i] = (i < 512) ? bq[i] : (i < 1024) ? bk[i - 512] : bv[i - 1024];
        return;
    }
    i -= QKV_COLS;
    if (i < 4096) {   // Wm: 4 ct * 2 kk * 32 lanes * 16
        int j = i & 15, l = (i >> 4) & 31, kk = (i >> 9) & 1, ct = i >> 10;
        int K = kk * 32 + ((l < 16) ? 0 : 16) + j;
        int e = ct * 16 + (l & 15);
        Bwm[i] = f2bf(Wm[K * OUT_C + e]);
    }
}

// ---------------------------------------------------------------------------
// Kernel 2: fused QKV GEMM (bf16 WMMA, f32 acc) + bias + fp() feature map.
// Each block: 32 rows. 8 waves x 12 column tiles; per (ct,kk) B fragment
// is loaded once and consumed by two 16-row WMMAs. Outputs phi_q, phi_k, v
// as bf16 [N, H*64] row-major.
// ---------------------------------------------------------------------------
__global__ void __launch_bounds__(256)
qkv_phi(const float* __restrict__ x,
        const unsigned short* __restrict__ Bqkv,
        const float* __restrict__ biasQKV,
        const float* __restrict__ nscale,
        unsigned short* __restrict__ phiQ,
        unsigned short* __restrict__ phiK,
        unsigned short* __restrict__ vOut)
{
    __shared__ __align__(16) unsigned short xt[ROWT][IN_C];       // 16 KB
    __shared__ __align__(16) unsigned short sQKV[ROWT][QKV_COLS]; // 96 KB (bf16)
    const int tid = threadIdx.x;
    const long row0 = (long)blockIdx.x * ROWT;

    // stage 1: x tile -> bf16 LDS (A operands served from here)
    for (int idx = tid; idx < ROWT * IN_C; idx += 256) {
        int r = idx >> 8, c = idx & 255;
        xt[r][c] = f2bf(x[(row0 + r) * IN_C + c]);
    }
    __syncthreads();

    // stage 2: WMMA.  A 16x32 bf16 per-lane: elems 0..15 = K(+0..15 | +16..31)
    // depending on lane half.
    {
        const int wave = tid >> 5, lane = tid & 31;
        const int m    = lane & 15;
        const int koff = (lane < 16) ? 0 : 16;
        const int rb   = (lane < 16) ? 0 : 8;
        for (int t = 0; t < CT_QKV / 8; ++t) {
            int ct = wave + t * 8;
            v8f a0 = {0.f,0.f,0.f,0.f,0.f,0.f,0.f,0.f};
            v8f a1 = {0.f,0.f,0.f,0.f,0.f,0.f,0.f,0.f};
            for (int kk = 0; kk < KK_QKV; ++kk) {
                ABu A0, A1, B;
                const unsigned short* bp =
                    Bqkv + (((size_t)(ct * KK_QKV + kk) * 32 + lane) << 4);
                B.b[0] = *(const B128*)bp;
                B.b[1] = *(const B128*)(bp + 8);
                const unsigned short* ap0 = &xt[m][kk * 32 + koff];
                A0.b[0] = *(const B128*)ap0;
                A0.b[1] = *(const B128*)(ap0 + 8);
                const unsigned short* ap1 = &xt[m + 16][kk * 32 + koff];
                A1.b[0] = *(const B128*)ap1;
                A1.b[1] = *(const B128*)(ap1 + 8);
                a0 = __builtin_amdgcn_wmma_f32_16x16x32_bf16(
                         false, A0.v, false, B.v, (short)0, a0, false, false);
                a1 = __builtin_amdgcn_wmma_f32_16x16x32_bf16(
                         false, A1.v, false, B.v, (short)0, a1, false, false);
            }
            int col = ct * 16 + m;
            float bb = biasQKV[col];
            for (int i2 = 0; i2 < 8; ++i2) {          // C layout: VGPR i -> row rb+i
                sQKV[rb + i2][col]      = f2bf(a0[i2] + bb);
                sQKV[16 + rb + i2][col] = f2bf(a1[i2] + bb);
            }
        }
    }
    __syncthreads();

    // stage 3: fp() feature map, one thread per (row, head, q|k); 2 tasks/thread
    {
        float denom = fabsf(nscale[0]) + 1e-6f;
        for (int tt = 0; tt < 2; ++tt) {
            int task = tid + tt * 256;
            int r = task >> 4, p = task & 15;
            int h = p & 7, isK = p >> 3;
            int base = (isK ? 512 : 0) + h * 64;
            float s2 = 0.f, s4 = 0.f;
            for (int j = 0; j < 64; ++j) {
                float s = bf2f(sQKV[r][base + j]);
                float y = (fmaxf(s, 0.f) + 1e-6f) / denom;
                float y2 = y * y;
                s2 += y2; s4 += y2 * y2;
            }
            float scale = sqrtf(s2) / (sqrtf(s4) + 1e-12f);
            unsigned int* dst = (unsigned int*)((isK ? phiK : phiQ)
                                                + (size_t)(row0 + r) * 512 + h * 64);
            for (int j2 = 0; j2 < 32; ++j2) {
                float sA = bf2f(sQKV[r][base + 2 * j2]);
                float sB = bf2f(sQKV[r][base + 2 * j2 + 1]);
                float yA = (fmaxf(sA, 0.f) + 1e-6f) / denom;
                float yB = (fmaxf(sB, 0.f) + 1e-6f) / denom;
                unsigned short pa = f2bf(scale * yA * yA);
                unsigned short pb = f2bf(scale * yB * yB);
                dst[j2] = ((unsigned int)pb << 16) | pa;
            }
        }
    }
    // v part already bf16 in LDS -> straight copy out
    {
        unsigned int* vdst = (unsigned int*)(vOut + (size_t)row0 * 512);
        for (int idx = tid; idx < ROWT * 256; idx += 256) {
            int r = idx >> 8, c2 = idx & 255;
            unsigned short a = sQKV[r][1024 + 2 * c2];
            unsigned short b = sQKV[r][1024 + 2 * c2 + 1];
            vdst[(size_t)r * 256 + c2] = ((unsigned int)b << 16) | a;
        }
    }
}

__global__ void zero_f32(float* __restrict__ p, int n) {
    int i = blockIdx.x * 256 + threadIdx.x;
    if (i < n) p[i] = 0.f;
}

// ---------------------------------------------------------------------------
// Kernel 4: kT_v[h,m,d] = sum_n phi_k[n,h,m] * v[n,h,d] ; sum_k[h,m].
// grid = 8 heads x 64 chunks; thread -> (m, 16-wide d group); one atomic flush.
// ---------------------------------------------------------------------------
__global__ void __launch_bounds__(256)
ktv_reduce(const unsigned short* __restrict__ phiK,
           const unsigned short* __restrict__ vIn,
           float* __restrict__ kTv, float* __restrict__ sumk)
{
    const int h = blockIdx.x & 7;
    const int chunk = blockIdx.x >> 3;
    const int CH = (NROWS + 63) / 64;      // 1563
    long n0 = (long)chunk * CH;
    long n1 = n0 + CH; if (n1 > NROWS) n1 = NROWS;
    const int t  = threadIdx.x;
    const int mm = t >> 2;
    const int d0 = (t & 3) * 16;
    float acc[16];
    for (int j = 0; j < 16; ++j) acc[j] = 0.f;
    float accS = 0.f;
    for (long n = n0; n < n1; ++n) {
        float pk = bf2f(phiK[n * 512 + h * 64 + mm]);
        const B128* vp = (const B128*)(vIn + n * 512 + h * 64 + d0);
        B128 q0 = vp[0], q1 = vp[1];
        const unsigned short* v0 = (const unsigned short*)&q0;
        const unsigned short* v1 = (const unsigned short*)&q1;
        for (int j = 0; j < 8; ++j) acc[j]     += pk * bf2f(v0[j]);
        for (int j = 0; j < 8; ++j) acc[8 + j] += pk * bf2f(v1[j]);
        if ((t & 3) == 0) accS += pk;
    }
    float* dst = kTv + ((size_t)(h * 64 + mm)) * 64 + d0;
    for (int j = 0; j < 16; ++j) atomicAdd(&dst[j], acc[j]);
    if ((t & 3) == 0) atomicAdd(&sumk[h * 64 + mm], accS);
}

// Kernel 5: kT_v f32 -> bf16 WMMA B layout [h][ct][kk][lane][16]
__global__ void ktv_pack(const float* __restrict__ kTv,
                         unsigned short* __restrict__ Bktv)
{
    int i = blockIdx.x * 256 + threadIdx.x;
    if (i >= NH * 4096) return;
    int j = i & 15, l = (i >> 4) & 31, kk = (i >> 9) & 1, ct = (i >> 10) & 3, h = i >> 12;
    int K = kk * 32 + ((l < 16) ? 0 : 16) + j;     // m index (K dim)
    int e = ct * 16 + (l & 15);                    // d index (N dim)
    Bktv[i] = f2bf(kTv[((size_t)(h * 64 + K)) * 64 + e]);
}

// ---------------------------------------------------------------------------
// Kernel 6: numerator (phi_q @ kT_v) + residual (v @ Wm) via WMMA,
// denominator dot, head mean, Lorentz lift.  32 rows/block; each Bktv/Bwm
// fragment feeds 2 row sub-tiles (4 WMMAs per kk iteration).
// ---------------------------------------------------------------------------
__global__ void __launch_bounds__(256)
attn_out(const unsigned short* __restrict__ phiQ,
         const unsigned short* __restrict__ vIn,
         const unsigned short* __restrict__ Bktv,
         const unsigned short* __restrict__ Bwm,
         const float* __restrict__ sumk,
         const float* __restrict__ bm,
         float* __restrict__ out)
{
    __shared__ __align__(16) unsigned short pq[ROWT][512];   // 32 KB
    __shared__ __align__(16) unsigned short vt[ROWT][512];   // 32 KB
    __shared__ float denomS[ROWT][NH];                       // 1 KB
    __shared__ float osum[ROWT][OUT_C];                      // 8 KB
    const int tid = threadIdx.x;
    const long row0 = (long)blockIdx.x * ROWT;

    {
        const unsigned int* pqs = (const unsigned int*)(phiQ + (size_t)row0 * 512);
        const unsigned int* vts = (const unsigned int*)(vIn  + (size_t)row0 * 512);
        unsigned int* pqd = (unsigned int*)&pq[0][0];
        unsigned int* vtd = (unsigned int*)&vt[0][0];
        for (int idx = tid; idx < ROWT * 256; idx += 256) {
            pqd[idx] = pqs[idx];
            vtd[idx] = vts[idx];
        }
    }
    for (int idx = tid; idx < ROWT * OUT_C; idx += 256) (&osum[0][0])[idx] = 0.f;
    __syncthreads();

    {   // denominators: phi_q . sum_k per (row, head); exactly 256 tasks
        int r = tid >> 3, h = tid & 7;
        float s = 0.f;
        for (int j = 0; j < 64; ++j)
            s += bf2f(pq[r][h * 64 + j]) * sumk[h * 64 + j];
        denomS[r][h] = s;
    }
    __syncthreads();

    {   // wave w == head w: 4 column tiles x (K=64 as 2x 16x16x32), num + residual
        const int h = tid >> 5, lane = tid & 31;
        const int m    = lane & 15;
        const int koff = (lane < 16) ? 0 : 16;
        const int rb   = (lane < 16) ? 0 : 8;
        for (int ct = 0; ct < 4; ++ct) {
            v8f n0 = {0.f,0.f,0.f,0.f,0.f,0.f,0.f,0.f};
            v8f n1 = {0.f,0.f,0.f,0.f,0.f,0.f,0.f,0.f};
            v8f r0 = {0.f,0.f,0.f,0.f,0.f,0.f,0.f,0.f};
            v8f r1 = {0.f,0.f,0.f,0.f,0.f,0.f,0.f,0.f};
            for (int kk = 0; kk < 2; ++kk) {
                ABu Aq0, Aq1, Av0, Av1, Bk, Bw;
                int abase = h * 64 + kk * 32 + koff;
                const unsigned short* bk_ =
                    Bktv + (size_t)h * 4096 + (((ct * 2 + kk) * 32 + lane) << 4);
                Bk.b[0] = *(const B128*)bk_; Bk.b[1] = *(const B128*)(bk_ + 8);
                const unsigned short* bw =
                    Bwm + (((ct * 2 + kk) * 32 + lane) << 4);
                Bw.b[0] = *(const B128*)bw;  Bw.b[1] = *(const B128*)(bw + 8);
                const unsigned short* aq0 = &pq[m][abase];
                Aq0.b[0] = *(const B128*)aq0; Aq0.b[1] = *(const B128*)(aq0 + 8);
                const unsigned short* aq1 = &pq[m + 16][abase];
                Aq1.b[0] = *(const B128*)aq1; Aq1.b[1] = *(const B128*)(aq1 + 8);
                const unsigned short* av0 = &vt[m][abase];
                Av0.b[0] = *(const B128*)av0; Av0.b[1] = *(const B128*)(av0 + 8);
                const unsigned short* av1 = &vt[m + 16][abase];
                Av1.b[0] = *(const B128*)av1; Av1.b[1] = *(const B128*)(av1 + 8);
                n0 = __builtin_amdgcn_wmma_f32_16x16x32_bf16(
                         false, Aq0.v, false, Bk.v, (short)0, n0, false, false);
                n1 = __builtin_amdgcn_wmma_f32_16x16x32_bf16(
                         false, Aq1.v, false, Bk.v, (short)0, n1, false, false);
                r0 = __builtin_amdgcn_wmma_f32_16x16x32_bf16(
                         false, Av0.v, false, Bw.v, (short)0, r0, false, false);
                r1 = __builtin_amdgcn_wmma_f32_16x16x32_bf16(
                         false, Av1.v, false, Bw.v, (short)0, r1, false, false);
            }
            int col = ct * 16 + m;
            float bmv = bm[col];
            for (int i2 = 0; i2 < 8; ++i2) {
                int ra = rb + i2, rbig = 16 + rb + i2;
                float v0 = n0[i2] / (denomS[ra][h]   + 1e-6f) + r0[i2] + bmv;
                float v1 = n1[i2] / (denomS[rbig][h] + 1e-6f) + r1[i2] + bmv;
                atomicAdd(&osum[ra][col],   v0);   // LDS atomic: sum over heads
                atomicAdd(&osum[rbig][col], v1);
            }
        }
    }
    __syncthreads();

    for (int idx = tid; idx < ROWT * OUT_C; idx += 256) {  // head mean + space part
        int r = idx >> 6, c = idx & 63;
        float mval = osum[r][c] * (1.0f / NH);
        osum[r][c] = mval;
        out[(row0 + r) * 65 + 1 + c] = mval;
    }
    __syncthreads();
    if (tid < ROWT) {                                      // Lorentz lift, K_CURV=1
        float ss = 0.f;
        for (int c = 0; c < OUT_C; ++c) { float v2 = osum[tid][c]; ss += v2 * v2; }
        out[(row0 + tid) * 65] = sqrtf(ss + 1.0f);
    }
}

// ---------------------------------------------------------------------------
// Workspace layout (bytes), all 16B aligned:
//   phiQ  [N*512 bf16]   @ 0           102,400,000
//   phiK  [N*512 bf16]   @ 102,400,000 102,400,000
//   vB    [N*512 bf16]   @ 204,800,000 102,400,000
//   Bqkv  [393216 bf16]  @ 307,200,000     786,432
//   bias  [1536 f32]     @ 307,986,432       6,144
//   Bwm   [4096 bf16]    @ 307,992,576       8,192
//   kTv   [32768 f32]    @ 308,000,768     131,072
//   sumk  [512 f32]      @ 308,131,840       2,048   (contiguous w/ kTv)
//   Bktv  [32768 bf16]   @ 308,133,888      65,536
// ---------------------------------------------------------------------------
extern "C" void kernel_launch(void* const* d_in, const int* in_sizes, int n_in,
                              void* d_out, int out_size, void* d_ws, size_t ws_size,
                              hipStream_t stream) {
    (void)in_sizes; (void)n_in; (void)out_size; (void)ws_size;
    const float* x      = (const float*)d_in[0];
    const float* Wq     = (const float*)d_in[1];
    const float* bq     = (const float*)d_in[2];
    const float* Wk     = (const float*)d_in[3];
    const float* bk     = (const float*)d_in[4];
    const float* Wv     = (const float*)d_in[5];
    const float* bv     = (const float*)d_in[6];
    const float* nscale = (const float*)d_in[7];
    const float* Wm     = (const float*)d_in[8];
    const float* bm     = (const float*)d_in[9];
    float* out = (float*)d_out;
    char* ws = (char*)d_ws;

    unsigned short* phiQ = (unsigned short*)(ws + 0);
    unsigned short* phiK = (unsigned short*)(ws + 102400000ull);
    unsigned short* vB   = (unsigned short*)(ws + 204800000ull);
    unsigned short* Bqkv = (unsigned short*)(ws + 307200000ull);
    float*          bias = (float*)         (ws + 307986432ull);
    unsigned short* Bwm  = (unsigned short*)(ws + 307992576ull);
    float*          kTv  = (float*)         (ws + 308000768ull);
    float*          sumk = (float*)         (ws + 308131840ull);
    unsigned short* Bktv = (unsigned short*)(ws + 308133888ull);

    const int prepTasks = CT_QKV * KK_QKV * 32 * 16 + QKV_COLS + 4096; // 398848
    prep_pack<<<(prepTasks + 255) / 256, 256, 0, stream>>>(
        Wq, bq, Wk, bk, Wv, bv, Wm, Bqkv, bias, Bwm);

    qkv_phi<<<NROWS / ROWT, 256, 0, stream>>>(
        x, Bqkv, bias, nscale, phiQ, phiK, vB);

    zero_f32<<<(33280 + 255) / 256, 256, 0, stream>>>(kTv, 33280); // kTv + sumk

    ktv_reduce<<<NH * 64, 256, 0, stream>>>(phiK, vB, kTv, sumk);

    ktv_pack<<<(NH * 4096 + 255) / 256, 256, 0, stream>>>(kTv, Bktv);

    attn_out<<<NROWS / ROWT, 256, 0, stream>>>(
        phiQ, vB, Bktv, Bwm, sumk, bm, out);
}
